// PaiNNBlock_60601988547146
// MI455X (gfx1250) — matrix operand order repaired and checked by reading, not compiled, not run on hardware
//
#include <hip/hip_runtime.h>
#include <cmath>

// ---------------- constants -----------------------------------------------
constexpr int NN = 10000;    // nodes
constexpr int EE = 320000;   // edges
constexpr int HD = 128;      // hidden
constexpr int RR = 20;       // rbf
constexpr int H3 = 384;      // 3*H

typedef _Float16 half_t;
typedef __attribute__((ext_vector_type(16))) _Float16 v16h;
typedef __attribute__((ext_vector_type(8)))  _Float16 h8;
typedef __attribute__((ext_vector_type(4)))  _Float16 h4;
typedef __attribute__((ext_vector_type(8)))  float    v8f;
typedef __attribute__((ext_vector_type(4)))  float    f4;

// ---------------- WMMA helpers --------------------------------------------
__device__ inline v8f wmma16(v16h a, v16h b, v8f c) {
  // D = A(16x32 f16) * B(32x16 f16) + C(16x16 f32)
  return __builtin_amdgcn_wmma_f32_16x16x32_f16(
      /*neg_a=*/false, a, /*neg_b=*/false, b,
      /*c_mod=*/(short)0, c, /*reuse_a=*/false, /*reuse_b=*/false);
}

// fast silu: one v_exp + one v_rcp (precision irrelevant at f16 msg precision)
__device__ inline float silu_f(float x) {
  return x * __builtin_amdgcn_rcpf(1.0f + __expf(-x));
}

// A fragment (16x32 f16) from an f16 row (global or LDS): 2x b128 loads.
__device__ inline v16h load_a_f16(const half_t* row, int kbase) {
  int khalf = (threadIdx.x >> 4) & 1;
  h8 lo = *(const h8*)(row + kbase + khalf * 8);
  h8 hi = *(const h8*)(row + kbase + 16 + khalf * 8);
  return __builtin_shufflevector(lo, hi, 0, 1, 2, 3, 4, 5, 6, 7,
                                 8, 9, 10, 11, 12, 13, 14, 15);
}

// B fragment from pre-swizzled weights: one contiguous 32B v16h per lane.
__device__ inline v16h load_b_swz(const half_t* swz, int Kt, int ntile, int kt) {
  int lane = threadIdx.x & 31;
  const v16h* p = (const v16h*)swz;
  return p[(size_t)((ntile * Kt + kt) * 32 + lane)];
}

// ---------------- pre-pass kernels -----------------------------------------
// Weight swizzle: [ntile][kt][lane][16 halfs]; K = kt*32 + (idx<8 ? (lane>>4)*8 + idx : 16 + (lane>>4)*8 + idx-8).
__global__ void swizzle_b_kernel(const float* __restrict__ w, half_t* __restrict__ dst,
                                 int K, int Kvalid, int Ncols) {
  int i = blockIdx.x * 256 + threadIdx.x;
  int total = K * Ncols;
  if (i >= total) return;
  int idx  = i & 15;
  int lane = (i >> 4) & 31;
  int tk   = i >> 9;
  int Kt   = K >> 5;
  int kt    = tk % Kt;
  int ntile = tk / Kt;
  int khalf = lane >> 4;
  int n     = ntile * 16 + (lane & 15);
  int kk    = kt * 32 + (idx < 8 ? khalf * 8 + idx : 16 + khalf * 8 + (idx - 8));
  float v = (kk < Kvalid) ? w[(size_t)kk * Ncols + n] : 0.0f;
  dst[i] = (_Float16)v;
}

__global__ void copy_f32_kernel(const float* __restrict__ src, float* __restrict__ dst, int n) {
  int i = blockIdx.x * 256 + threadIdx.x;
  if (i < n) dst[i] = src[i];
}

__global__ void cvt_f16_kernel(const float* __restrict__ src, half_t* __restrict__ dst, int n) {
  int i = blockIdx.x * 256 + threadIdx.x;
  if (i < n) dst[i] = (_Float16)src[i];
}

// rbf [E,20] f32 -> [E,32] f16, zero padded (branch-free edge-kernel loads)
__global__ void pad_rbf_kernel(const float* __restrict__ rbf, half_t* __restrict__ dst) {
  int i = blockIdx.x * 256 + threadIdx.x;
  if (i >= EE * 32) return;
  int e = i >> 5, c = i & 31;
  dst[i] = (c < RR) ? (_Float16)rbf[(size_t)e * RR + c] : (_Float16)0.0f;
}

// ---------------- edge (interaction) kernel --------------------------------
// 32 edges per block (2 M-tiles), 8 waves. Wave w owns h-slice [16w,16w+16);
// each B fragment is reused across both M-tiles.
__global__ __launch_bounds__(256) void edge_kernel(
    const half_t* __restrict__ sh, const float* __restrict__ v,
    const long long* __restrict__ ei, const half_t* __restrict__ rbfh,
    const float* __restrict__ unit,
    const half_t* __restrict__ fw1z, const float* __restrict__ fb1,
    const half_t* __restrict__ fw2z, const float* __restrict__ fb2,
    const half_t* __restrict__ sw1z, const float* __restrict__ sb1,
    const half_t* __restrict__ sw2z, const float* __restrict__ sb2,
    float* __restrict__ s_mid, float* __restrict__ v_mid) {
  constexpr int LDP = HD + 8;  // padded row stride (halfs); 272B = 17*16B aligned
  __shared__ half_t X1s[32][LDP];
  __shared__ half_t Y1s[32][LDP];
  __shared__ int   srcS[32];
  __shared__ int   dstS[32];
  __shared__ float unitS[32][3];

  const int tid   = threadIdx.x;
  const int w     = tid >> 5;
  const int lane  = tid & 31;
  const int m     = lane & 15;
  const int khalf = lane >> 4;
  const int base  = blockIdx.x * 32;

  if (tid < 32) {
    int e = base + tid;
    srcS[tid] = (int)ei[e];
    dstS[tid] = (int)ei[EE + e];
    unitS[tid][0] = unit[(size_t)e * 3 + 0];
    unitS[tid][1] = unit[(size_t)e * 3 + 1];
    unitS[tid][2] = unit[(size_t)e * 3 + 2];
  }
  __syncthreads();

  const v8f vzero = {0.f, 0.f, 0.f, 0.f, 0.f, 0.f, 0.f, 0.f};
  const int hcol = 16 * w + m;

  // ---- Stage 1: X1 = silu(rbf@fw1+fb1), Y1 = silu(s[src]@sw1+sb1), 2 M-tiles
  {
    const v16h bF = load_b_swz(fw1z, /*Kt=*/1, /*ntile=*/w, /*kt=*/0);
    v16h bS[4];
#pragma unroll
    for (int kt = 0; kt < 4; ++kt) bS[kt] = load_b_swz(sw1z, 4, w, kt);
    const float bias1 = fb1[hcol];
    const float bias2 = sb1[hcol];

#pragma unroll
    for (int mt = 0; mt < 2; ++mt) {
      const int e = base + mt * 16 + m;
      // filter first layer (K padded 20->32): 1 WMMA, clean b128 loads
      v8f c = vzero;
      c = wmma16(load_a_f16(rbfh + (size_t)e * 32, 0), bF, c);
#pragma unroll
      for (int r = 0; r < 8; ++r)
        X1s[mt * 16 + r + 8 * khalf][hcol] = (_Float16)silu_f(c[r] + bias1);

      // scalar first layer: K=128 -> 4 WMMA, f16 gather of s[src]
      const half_t* srow = sh + (size_t)srcS[mt * 16 + m] * HD;
      v8f c2 = vzero;
#pragma unroll
      for (int kt = 0; kt < 4; ++kt)
        c2 = wmma16(load_a_f16(srow, kt * 32), bS[kt], c2);
#pragma unroll
      for (int r = 0; r < 8; ++r)
        Y1s[mt * 16 + r + 8 * khalf][hcol] = (_Float16)silu_f(c2[r] + bias2);
    }
  }
  __syncthreads();

  // ---- Stage 2: second layers; 6 B fragments/kt, each reused for 2 M-tiles
  v8f cF[2][3] = {{vzero, vzero, vzero}, {vzero, vzero, vzero}};
  v8f cS[2][3] = {{vzero, vzero, vzero}, {vzero, vzero, vzero}};
  const half_t* Xrow0 = &X1s[m][0];
  const half_t* Xrow1 = &X1s[16 + m][0];
  const half_t* Yrow0 = &Y1s[m][0];
  const half_t* Yrow1 = &Y1s[16 + m][0];
#pragma unroll
  for (int kt = 0; kt < 4; ++kt) {
    v16h ax0 = load_a_f16(Xrow0, kt * 32);
    v16h ax1 = load_a_f16(Xrow1, kt * 32);
    v16h ay0 = load_a_f16(Yrow0, kt * 32);
    v16h ay1 = load_a_f16(Yrow1, kt * 32);
#pragma unroll
    for (int t = 0; t < 3; ++t) {
      v16h bf = load_b_swz(fw2z, 4, w + 8 * t, kt);
      cF[0][t] = wmma16(ax0, bf, cF[0][t]);
      cF[1][t] = wmma16(ax1, bf, cF[1][t]);
      v16h bs = load_b_swz(sw2z, 4, w + 8 * t, kt);
      cS[0][t] = wmma16(ay0, bs, cS[0][t]);
      cS[1][t] = wmma16(ay1, bs, cS[1][t]);
    }
  }

  const float fb0 = fb2[hcol], fb1v = fb2[HD + hcol], fb2v = fb2[2 * HD + hcol];
  const float sb0 = sb2[hcol], sb1v = sb2[HD + hcol], sb2v = sb2[2 * HD + hcol];

#pragma unroll
  for (int mt = 0; mt < 2; ++mt) {
#pragma unroll
    for (int r = 0; r < 8; ++r) {
      int mr = mt * 16 + r + 8 * khalf;   // edge within block tile
      int dn = dstS[mr];
      int sn = srcS[mr];
      float msDS = (cF[mt][0][r] + fb0)  * (cS[mt][0][r] + sb0);
      float msVV = (cF[mt][1][r] + fb1v) * (cS[mt][1][r] + sb1v);
      float msVR = (cF[mt][2][r] + fb2v) * (cS[mt][2][r] + sb2v);
      atomicAdd(&s_mid[(size_t)dn * HD + hcol], msDS);
#pragma unroll
      for (int ax3 = 0; ax3 < 3; ++ax3) {
        float vsrc = v[((size_t)sn * 3 + ax3) * HD + hcol];
        float val  = msVV * vsrc + msVR * unitS[mr][ax3];
        atomicAdd(&v_mid[((size_t)dn * 3 + ax3) * HD + hcol], val);
      }
    }
  }
}

// ---------------- node (update) kernel --------------------------------------
// 16 nodes per block (48 v-rows), 8 waves.
__global__ __launch_bounds__(256) void update_kernel(
    const float* __restrict__ s_mid, const float* __restrict__ v_mid,
    const half_t* __restrict__ Uz, const half_t* __restrict__ Vz,
    const half_t* __restrict__ uw1z, const float* __restrict__ ub1,
    const half_t* __restrict__ uw2z, const float* __restrict__ ub2,
    float* __restrict__ out_s, float* __restrict__ out_v) {
  constexpr int LDP  = HD + 8;
  constexpr int LDP2 = 2 * HD + 8;
  __shared__ half_t vmh[48][LDP];
  __shared__ half_t vuh[48][LDP];
  __shared__ half_t vvh[48][LDP];
  __shared__ float  innerS[16][HD];
  __shared__ half_t A2h[16][LDP2];
  __shared__ half_t G1h[16][LDP];

  const int tid   = threadIdx.x;
  const int w     = tid >> 5;
  const int lane  = tid & 31;
  const int mlo   = lane & 15;
  const int khalf = lane >> 4;
  const int n0    = blockIdx.x * 16;
  const v8f vzero = {0.f, 0.f, 0.f, 0.f, 0.f, 0.f, 0.f, 0.f};

  // (a) stage v rows (local row = i*3+axis) into LDS as f16, vectorized
  for (int idx = tid; idx < 48 * (HD / 4); idx += 256) {
    int row = idx / (HD / 4);
    int c4  = (idx % (HD / 4)) * 4;
    f4 x = *(const f4*)&v_mid[((size_t)n0 * 3 + row) * HD + c4];
    h4 y;
#pragma unroll
    for (int i = 0; i < 4; ++i) y[i] = (_Float16)x[i];
    *(h4*)&vmh[row][c4] = y;
  }
  __syncthreads();

  // (b) v_u = v@U, v_v = v@V : 48 tiles, 6 per wave (wave owns ntile=w)
  for (int t = w; t < 48; t += 8) {
    int mat = t / 24;
    int rem = t % 24;
    int mt  = rem / 8;
    int nt  = rem % 8;
    const half_t* arow = &vmh[mt * 16 + mlo][0];
    const half_t* B = mat ? Vz : Uz;
    v8f c = vzero;
#pragma unroll
    for (int kt = 0; kt < 4; ++kt)
      c = wmma16(load_a_f16(arow, kt * 32), load_b_swz(B, 4, nt, kt), c);
#pragma unroll
    for (int r = 0; r < 8; ++r) {
      int row = mt * 16 + r + 8 * khalf;
      int col = nt * 16 + mlo;
      if (mat) vvh[row][col] = (_Float16)c[r];
      else     vuh[row][col] = (_Float16)c[r];
    }
  }
  __syncthreads();

  // (c) inner product, vector norm, build A2 = [s | ||v_v||]
  for (int idx = tid; idx < 16 * HD; idx += 256) {
    int i = idx / HD, h = idx % HD;
    float acc = 0.f, nrm = 0.f;
#pragma unroll
    for (int ax = 0; ax < 3; ++ax) {
      float a1 = (float)vuh[i * 3 + ax][h];
      float b1 = (float)vvh[i * 3 + ax][h];
      acc += a1 * b1;
      nrm += b1 * b1;
    }
    innerS[i][h] = acc;
    A2h[i][h]      = (_Float16)s_mid[(size_t)(n0 + i) * HD + h];
    A2h[i][HD + h] = (_Float16)sqrtf(nrm);
  }
  __syncthreads();

  // (d) G1 = silu(A2 @ uw1 + ub1) : K=256 -> 8 WMMA per wave (ntile = w)
  {
    const half_t* arow = &A2h[mlo][0];
    v8f c = vzero;
#pragma unroll
    for (int kt = 0; kt < 8; ++kt)
      c = wmma16(load_a_f16(arow, kt * 32), load_b_swz(uw1z, 8, w, kt), c);
    float bias = ub1[16 * w + mlo];
#pragma unroll
    for (int r = 0; r < 8; ++r)
      G1h[r + 8 * khalf][16 * w + mlo] = (_Float16)silu_f(c[r] + bias);
  }
  __syncthreads();

  // (e) upd = G1 @ uw2 + ub2 ; fused epilogue
  {
    const half_t* arow = &G1h[mlo][0];
    v8f cA = vzero, cB = vzero, cC = vzero;
#pragma unroll
    for (int kt = 0; kt < 4; ++kt) {
      v16h a = load_a_f16(arow, kt * 32);
      cA = wmma16(a, load_b_swz(uw2z, 4, w,      kt), cA);
      cB = wmma16(a, load_b_swz(uw2z, 4, w + 8,  kt), cB);
      cC = wmma16(a, load_b_swz(uw2z, 4, w + 16, kt), cC);
    }
    const int hcol = 16 * w + mlo;
    const float ba = ub2[hcol], bb = ub2[HD + hcol], bc = ub2[2 * HD + hcol];
#pragma unroll
    for (int r = 0; r < 8; ++r) {
      int il = r + 8 * khalf;
      int node = n0 + il;
      float aV = cA[r] + ba;
      float bV = cB[r] + bb;
      float cV = cC[r] + bc;
      float sOld = s_mid[(size_t)node * HD + hcol];
      out_s[(size_t)node * HD + hcol] = sOld + aV + bV * innerS[il][hcol];
#pragma unroll
      for (int ax = 0; ax < 3; ++ax) {
        float vm = v_mid[((size_t)node * 3 + ax) * HD + hcol];
        float vu = (float)vuh[il * 3 + ax][hcol];
        out_v[((size_t)node * 3 + ax) * HD + hcol] = vm + cV * vu;
      }
    }
  }
}

// ---------------- launch ----------------------------------------------------
extern "C" void kernel_launch(void* const* d_in, const int* in_sizes, int n_in,
                              void* d_out, int out_size, void* d_ws, size_t ws_size,
                              hipStream_t stream) {
  const float*     s    = (const float*)d_in[0];
  const float*     v    = (const float*)d_in[1];
  const long long* ei   = (const long long*)d_in[2];
  const float*     rbf  = (const float*)d_in[3];
  const float*     unit = (const float*)d_in[4];
  const float* fw1 = (const float*)d_in[5];
  const float* fb1 = (const float*)d_in[6];
  const float* fw2 = (const float*)d_in[7];
  const float* fb2 = (const float*)d_in[8];
  const float* sw1 = (const float*)d_in[9];
  const float* sb1 = (const float*)d_in[10];
  const float* sw2 = (const float*)d_in[11];
  const float* sb2 = (const float*)d_in[12];
  const float* U   = (const float*)d_in[13];
  const float* V   = (const float*)d_in[14];
  const float* uw1 = (const float*)d_in[15];
  const float* ub1 = (const float*)d_in[16];
  const float* uw2 = (const float*)d_in[17];
  const float* ub2 = (const float*)d_in[18];

  size_t off = 0;
  auto carve = [&](size_t bytes) -> void* {
    void* p = (char*)d_ws + off;
    off += (bytes + 255) & ~(size_t)255;
    return p;
  };
  float*  s_mid = (float*)carve((size_t)NN * HD * 4);
  float*  v_mid = (float*)carve((size_t)NN * 3 * HD * 4);
  half_t* sh    = (half_t*)carve((size_t)NN * HD * 2);
  half_t* rbfh  = (half_t*)carve((size_t)EE * 32 * 2);
  half_t* fw1z  = (half_t*)carve((size_t)32 * HD * 2);
  half_t* sw1z  = (half_t*)carve((size_t)HD * HD * 2);
  half_t* fw2z  = (half_t*)carve((size_t)HD * H3 * 2);
  half_t* sw2z  = (half_t*)carve((size_t)HD * H3 * 2);
  half_t* Uz    = (half_t*)carve((size_t)HD * HD * 2);
  half_t* Vz    = (half_t*)carve((size_t)HD * HD * 2);
  half_t* uw1z  = (half_t*)carve((size_t)2 * HD * HD * 2);
  half_t* uw2z  = (half_t*)carve((size_t)HD * H3 * 2);

  auto gsz = [](int n) { return (n + 255) / 256; };

  // weight swizzle (f16, WMMA B-fragment order)
  swizzle_b_kernel<<<gsz(32 * HD), 256, 0, stream>>>(fw1, fw1z, 32, RR, HD);
  swizzle_b_kernel<<<gsz(HD * HD), 256, 0, stream>>>(sw1, sw1z, HD, HD, HD);
  swizzle_b_kernel<<<gsz(HD * H3), 256, 0, stream>>>(fw2, fw2z, HD, HD, H3);
  swizzle_b_kernel<<<gsz(HD * H3), 256, 0, stream>>>(sw2, sw2z, HD, HD, H3);
  swizzle_b_kernel<<<gsz(HD * HD), 256, 0, stream>>>(U,   Uz,   HD, HD, HD);
  swizzle_b_kernel<<<gsz(HD * HD), 256, 0, stream>>>(V,   Vz,   HD, HD, HD);
  swizzle_b_kernel<<<gsz(2 * HD * HD), 256, 0, stream>>>(uw1, uw1z, 2 * HD, 2 * HD, HD);
  swizzle_b_kernel<<<gsz(HD * H3), 256, 0, stream>>>(uw2, uw2z, HD, HD, H3);

  // activations: residual init + f16 staging
  copy_f32_kernel<<<gsz(NN * HD), 256, 0, stream>>>(s, s_mid, NN * HD);
  copy_f32_kernel<<<gsz(NN * 3 * HD), 256, 0, stream>>>(v, v_mid, NN * 3 * HD);
  cvt_f16_kernel<<<gsz(NN * HD), 256, 0, stream>>>(s, sh, NN * HD);
  pad_rbf_kernel<<<gsz(EE * 32), 256, 0, stream>>>(rbf, rbfh);

  // interaction phase: 32 edges / block
  edge_kernel<<<EE / 32, 256, 0, stream>>>(sh, v, ei, rbfh, unit,
                                           fw1z, fb1, fw2z, fb2,
                                           sw1z, sb1, sw2z, sb2,
                                           s_mid, v_mid);

  // update phase: 16 nodes / block
  float* out_s = (float*)d_out;
  float* out_v = out_s + (size_t)NN * HD;
  update_kernel<<<NN / 16, 256, 0, stream>>>(s_mid, v_mid, Uz, Vz,
                                             uw1z, ub1, uw2z, ub2,
                                             out_s, out_v);
}
